// OCR_71631464563395
// MI455X (gfx1250) — compile-verified, matrix operand order checked by reading, not run on hardware
//
#include <hip/hip_runtime.h>
#include <hip/hip_bf16.h>

// OCR head for MI455X (gfx1250, wave32, WMMA).
// Heavy GEMMs (3x3 conv 2048->512, psi 512->256, rho 256->512, g 1024->512)
// run as f16 v_wmma_f32_16x16x32_f16 tiled GEMMs with double-buffered LDS
// (one barrier per k-chunk), async global->LDS weight copies (ASYNCcnt)
// pipelined one chunk ahead, and 2x ds_load_b128 per fragment per lane.

typedef __attribute__((ext_vector_type(16))) _Float16 v16h;
typedef __attribute__((ext_vector_type(8)))  _Float16 v8h;
typedef __attribute__((ext_vector_type(2)))  _Float16 v2h;
typedef __attribute__((ext_vector_type(8)))  float    v8f;
typedef int v4i_t __attribute__((vector_size(16)));

#define NC 19
#define APITCH 40  // f16 per LDS tile row: 80B rows -> 16B aligned, 20-dword stride (conflict-free)

#if __has_builtin(__builtin_amdgcn_global_load_async_to_lds_b128)
#define HAVE_ASYNC 1
#else
#define HAVE_ASYNC 0
#endif

__device__ __forceinline__ v8f wzero() {
  v8f z;
#pragma unroll
  for (int i = 0; i < 8; ++i) z[i] = 0.f;
  return z;
}

// XOR swizzle of the 8-element k-group inside a B row: breaks the bank
// conflict across the n-subtiles during transpose-scatter.
__device__ __forceinline__ int bswz(int n, int g) { return ((g ^ ((n >> 4) & 3)) << 3); }

// A fragment (16x32, rows = m): lane r holds row r; K hh..hh+7 and 16+hh..23+hh.
// LDS layout As[m][k] -> two contiguous 16B reads per lane.
__device__ __forceinline__ v16h frag_a(const _Float16* As, int m, int hh) {
  const _Float16* row = As + m * APITCH;
  v8h lo = *(const v8h*)(row + hh);
  v8h hi = *(const v8h*)(row + 16 + hh);
  v16h f;
#pragma unroll
  for (int i = 0; i < 8; ++i) { f[i] = lo[i]; f[i + 8] = hi[i]; }
  return f;
}

// B fragment (32x16, cols = n): lane r holds column r; same K split. LDS layout
// Bs[n][k] with group swizzle -> two contiguous 16B reads per lane.
__device__ __forceinline__ v16h frag_b(const _Float16* Bs, int n, int hh) {
  const _Float16* row = Bs + n * APITCH;
  const int g0 = hh >> 3;  // 0 or 1
  v8h lo = *(const v8h*)(row + bswz(n, g0));
  v8h hi = *(const v8h*)(row + bswz(n, g0 + 2));
  v16h f;
#pragma unroll
  for (int i = 0; i < 8; ++i) { f[i] = lo[i]; f[i + 8] = hi[i]; }
  return f;
}

// Copy 32B (16 f16) global -> LDS, async when the toolchain exposes the builtin.
__device__ __forceinline__ void copy32B_g2l(_Float16* dst, const _Float16* src) {
#if HAVE_ASYNC
  __builtin_amdgcn_global_load_async_to_lds_b128(
      (v4i_t*)src,
      (__attribute__((address_space(3))) v4i_t*)dst, 0, 0);
  __builtin_amdgcn_global_load_async_to_lds_b128(
      (v4i_t*)(src + 8),
      (__attribute__((address_space(3))) v4i_t*)(dst + 8), 0, 0);
#else
  reinterpret_cast<uint4*>(dst)[0] = reinterpret_cast<const uint4*>(src)[0];
  reinterpret_cast<uint4*>(dst)[1] = reinterpret_cast<const uint4*>(src)[1];
#endif
}

// ---------------------------------------------------------------------------
// Data-prep kernels
// ---------------------------------------------------------------------------

// x [4][2048][64][64] f32 -> xpad [4][2048][66][72] f16 (zero halo, rows padded
// to 72 so every row start is 16B aligned for vector loads)
__global__ __launch_bounds__(256) void pad_x_kernel(const float* __restrict__ x,
                                                    _Float16* __restrict__ xpad) {
  size_t i = (size_t)blockIdx.x * 256 + threadIdx.x;
  const size_t total = (size_t)4 * 2048 * 66 * 72;
  if (i >= total) return;
  int wp = (int)(i % 72);
  size_t r = i / 72;
  int hp = (int)(r % 66);
  size_t r2 = r / 66;
  int c = (int)(r2 % 2048);
  int b = (int)(r2 / 2048);
  float v = 0.f;
  if (hp >= 1 && hp <= 64 && wp >= 1 && wp <= 64)
    v = x[(((size_t)b * 2048 + c) * 64 + (hp - 1)) * 64 + (wp - 1)];
  xpad[i] = (_Float16)v;
}

// W_X [512][2048][3][3] f32 -> Wh [512][18432] f16, k = tap*2048 + c (k-contiguous)
__global__ __launch_bounds__(256) void wx_convert_kernel(const float* __restrict__ W,
                                                         _Float16* __restrict__ out) {
  size_t i = (size_t)blockIdx.x * 256 + threadIdx.x;
  if (i >= (size_t)512 * 18432) return;
  int k = (int)(i % 18432);
  int o = (int)(i / 18432);
  int tap = k >> 11;
  int c = k & 2047;
  out[i] = (_Float16)W[((size_t)o * 2048 + c) * 9 + tap];
}

// Plain f32 -> f16 cast (weights already [M][K] k-contiguous)
__global__ __launch_bounds__(256) void cast_f16_kernel(const float* __restrict__ W,
                                                       _Float16* __restrict__ out,
                                                       int n) {
  int i = blockIdx.x * 256 + threadIdx.x;
  if (i < n) out[i] = (_Float16)W[i];
}

// Transpose-scatter 16 n-contiguous f16 (shifted by uniform KX) into Bs[n][k].
template <int KX>
__device__ __forceinline__ void scatter_b16(_Float16* Bs, int no, int krow,
                                            v8h a, v8h b, v8h c) {
  const int gk = krow >> 3, wk = krow & 7;
#pragma unroll
  for (int j = 0; j < 16; ++j) {
    const int idx = j + KX;
    _Float16 v = (idx < 8) ? a[idx] : (idx < 16) ? b[idx - 8] : c[idx - 16];
    const int n = no + j;
    Bs[n * APITCH + bswz(n, gk) + wk] = v;
  }
}

// ---------------------------------------------------------------------------
// Implicit-GEMM 3x3 conv:  C[b][512][4096] = Wh @ im2col(xpad)
// Grid (16384/128, 512/128); 8 waves, each 32x64 = 2x4 WMMA fragments.
// Double-buffered LDS, one barrier per chunk; A-tiles via async global->LDS.
// ---------------------------------------------------------------------------
__global__ __launch_bounds__(256) void conv_wmma_kernel(const _Float16* __restrict__ Wh,
                                                        const _Float16* __restrict__ xpad,
                                                        float* __restrict__ C) {
  __shared__ _Float16 As[2][128 * APITCH];
  __shared__ _Float16 Bs[2][128 * APITCH];
  const int t = threadIdx.x;
  const int Nbase = blockIdx.x * 128;  // over b*h*w = 16384 (tile stays in one b)
  const int Mbase = blockIdx.y * 128;  // over 512 out channels
  const int lane = t & 31, wv = t >> 5;
  const int wm = wv >> 1, wn = wv & 1;
  const int r = lane & 15, hh = (lane >> 4) << 3;
  const int amr = t >> 1, ako = (t & 1) << 4;   // A: 2 thr/row, 16 k each
  const int bkr = t >> 3, bno = (t & 7) << 4;   // B: 8 thr/k-row, 16 n each
  const int n0 = Nbase + bno;
  const int bb = n0 >> 12, h0 = (n0 >> 6) & 63, w0 = n0 & 63;
  const _Float16* Arow = Wh + (size_t)(Mbase + amr) * 18432 + ako;

  v8f acc[2][4];
#pragma unroll
  for (int mi = 0; mi < 2; ++mi)
#pragma unroll
    for (int ni = 0; ni < 4; ++ni) acc[mi][ni] = wzero();

  // ---- prologue: stage chunk 0 (tap 0 -> ky=kx=0) ----
  v8h va, vb, vc;
  {
    const _Float16* brow = xpad + (((size_t)bb * 2048 + bkr) * 66 + h0) * 72 + w0;
    va = *(const v8h*)(brow);
    vb = *(const v8h*)(brow + 8);
    vc = *(const v8h*)(brow + 16);
  }
#if HAVE_ASYNC
  copy32B_g2l(&As[0][amr * APITCH + ako], Arow);
#else
  uint4 a0 = reinterpret_cast<const uint4*>(Arow)[0];
  uint4 a1 = reinterpret_cast<const uint4*>(Arow)[1];
#endif

  for (int kc = 0, it = 0; kc < 18432; kc += 32, ++it) {
    const int cur = it & 1;
    _Float16* Asb = &As[cur][0];
    _Float16* Bsb = &Bs[cur][0];
    // scatter staged B regs for this chunk (kx is uniform per chunk)
    const int tap = kc >> 11;
    const int kx = tap - (tap / 3) * 3;
    if (kx == 0)      scatter_b16<0>(Bsb, bno, bkr, va, vb, vc);
    else if (kx == 1) scatter_b16<1>(Bsb, bno, bkr, va, vb, vc);
    else              scatter_b16<2>(Bsb, bno, bkr, va, vb, vc);
#if HAVE_ASYNC
    __builtin_amdgcn_s_wait_asynccnt(0);  // my chunk-it A copy has landed
#else
    reinterpret_cast<uint4*>(&Asb[amr * APITCH + ako])[0] = a0;
    reinterpret_cast<uint4*>(&Asb[amr * APITCH + ako])[1] = a1;
#endif
    __syncthreads();  // single barrier: tiles of chunk `it` visible to all

    // stage chunk it+1 (overlaps with the WMMAs below)
    const int kcn = kc + 32;
    if (kcn < 18432) {
      const int tapn = kcn >> 11;
      const int kyn = tapn / 3;
      const int cchn = (kcn & 2047) + bkr;
      const _Float16* brow =
          xpad + (((size_t)bb * 2048 + cchn) * 66 + (h0 + kyn)) * 72 + w0;
      va = *(const v8h*)(brow);
      vb = *(const v8h*)(brow + 8);
      vc = *(const v8h*)(brow + 16);
      const _Float16* ag = Arow + kcn;
      __builtin_prefetch(ag + 32, 0, 1);
#if HAVE_ASYNC
      copy32B_g2l(&As[1 - cur][amr * APITCH + ako], ag);
#else
      a0 = reinterpret_cast<const uint4*>(ag)[0];
      a1 = reinterpret_cast<const uint4*>(ag)[1];
#endif
    }

    v16h af[2], bf[4];
#pragma unroll
    for (int mi = 0; mi < 2; ++mi) af[mi] = frag_a(Asb, wm * 32 + mi * 16 + r, hh);
#pragma unroll
    for (int ni = 0; ni < 4; ++ni) bf[ni] = frag_b(Bsb, wn * 64 + ni * 16 + r, hh);
#pragma unroll
    for (int mi = 0; mi < 2; ++mi)
#pragma unroll
      for (int ni = 0; ni < 4; ++ni)
        acc[mi][ni] = __builtin_amdgcn_wmma_f32_16x16x32_f16(
            false, af[mi], false, bf[ni], (short)0, acc[mi][ni], false, false);
  }
#pragma unroll
  for (int mi = 0; mi < 2; ++mi)
#pragma unroll
    for (int ni = 0; ni < 4; ++ni) {
      const int n = Nbase + wn * 64 + ni * 16 + r;
      const int bo = n >> 12, nl = n & 4095;
#pragma unroll
      for (int i = 0; i < 8; ++i) {
        const int m = Mbase + wm * 32 + mi * 16 + i + hh;
        C[((size_t)bo * 512 + m) * 4096 + nl] = acc[mi][ni][i];
      }
    }
}

// ---------------------------------------------------------------------------
// Generic WMMA GEMM: C[b][M][4096] = A[M][K1+K2] @ [B1;B2][b][K][4096]
// (split-K dual B source realizes the Xf||X_obj concat without materializing it)
// B tile: 8n x 2k per thread -> pair-packed ds_store_b32 transpose-scatter.
// ---------------------------------------------------------------------------
__global__ __launch_bounds__(256) void gemm_wmma_kernel(const _Float16* __restrict__ A,
                                                        const _Float16* __restrict__ B1,
                                                        const _Float16* __restrict__ B2,
                                                        float* __restrict__ C,
                                                        int M, int K1, int K2) {
  __shared__ _Float16 As[2][128 * APITCH];
  __shared__ _Float16 Bs[2][128 * APITCH];
  const int t = threadIdx.x;
  const int Nbase = blockIdx.x * 128;
  const int Mbase = blockIdx.y * 128;
  const int bb = blockIdx.z;
  const int lane = t & 31, wv = t >> 5;
  const int wm = wv >> 1, wn = wv & 1;
  const int r = lane & 15, hh = (lane >> 4) << 3;
  const int amr = t >> 1, ako = (t & 1) << 4;   // A: 2 thr/row, 16 k each
  const int bng = (t >> 4) << 3;                // B: n-group base (8 n)
  const int bkp = (t & 15) << 1;                // B: even k-pair 0..30
  const int Ktot = K1 + K2;
  const int nch = Ktot >> 5;
  const _Float16* Arow = A + (size_t)(Mbase + amr) * Ktot + ako;

  v8f acc[2][4];
#pragma unroll
  for (int mi = 0; mi < 2; ++mi)
#pragma unroll
    for (int ni = 0; ni < 4; ++ni) acc[mi][ni] = wzero();

  // stage chunk 0
  v8h blo, bhi;
  {
    const _Float16* bg = B1 + ((size_t)bb * K1 + bkp) * 4096 + Nbase + bng;
    blo = *(const v8h*)(bg);
    bhi = *(const v8h*)(bg + 4096);
  }
#if HAVE_ASYNC
  copy32B_g2l(&As[0][amr * APITCH + ako], Arow);
#else
  uint4 a0 = reinterpret_cast<const uint4*>(Arow)[0];
  uint4 a1 = reinterpret_cast<const uint4*>(Arow)[1];
#endif

  for (int it = 0; it < nch; ++it) {
    const int cur = it & 1;
    _Float16* Asb = &As[cur][0];
    _Float16* Bsb = &Bs[cur][0];
    // pair-packed transpose-scatter: 8 x ds_store_b32
    {
      const int gk = bkp >> 3, wk = bkp & 7;  // wk even
#pragma unroll
      for (int j = 0; j < 8; ++j) {
        const int n = bng + j;
        v2h pr;
        pr[0] = blo[j];
        pr[1] = bhi[j];
        *(v2h*)(Bsb + n * APITCH + bswz(n, gk) + wk) = pr;
      }
    }
#if HAVE_ASYNC
    __builtin_amdgcn_s_wait_asynccnt(0);
#else
    reinterpret_cast<uint4*>(&Asb[amr * APITCH + ako])[0] = a0;
    reinterpret_cast<uint4*>(&Asb[amr * APITCH + ako])[1] = a1;
#endif
    __syncthreads();

    // stage chunk it+1 (overlaps with WMMAs)
    if (it + 1 < nch) {
      const int kk = ((it + 1) << 5) + bkp;
      const _Float16* bg =
          (kk < K1) ? (B1 + ((size_t)bb * K1 + kk) * 4096 + Nbase + bng)
                    : (B2 + ((size_t)bb * K2 + (kk - K1)) * 4096 + Nbase + bng);
      blo = *(const v8h*)(bg);
      bhi = *(const v8h*)(bg + 4096);
      const _Float16* ag = Arow + ((it + 1) << 5);
      __builtin_prefetch(ag + 32, 0, 1);
#if HAVE_ASYNC
      copy32B_g2l(&As[1 - cur][amr * APITCH + ako], ag);
#else
      a0 = reinterpret_cast<const uint4*>(ag)[0];
      a1 = reinterpret_cast<const uint4*>(ag)[1];
#endif
    }

    v16h af[2], bf[4];
#pragma unroll
    for (int mi = 0; mi < 2; ++mi) af[mi] = frag_a(Asb, wm * 32 + mi * 16 + r, hh);
#pragma unroll
    for (int ni = 0; ni < 4; ++ni) bf[ni] = frag_b(Bsb, wn * 64 + ni * 16 + r, hh);
#pragma unroll
    for (int mi = 0; mi < 2; ++mi)
#pragma unroll
      for (int ni = 0; ni < 4; ++ni)
        acc[mi][ni] = __builtin_amdgcn_wmma_f32_16x16x32_f16(
            false, af[mi], false, bf[ni], (short)0, acc[mi][ni], false, false);
  }
#pragma unroll
  for (int mi = 0; mi < 2; ++mi)
#pragma unroll
    for (int ni = 0; ni < 4; ++ni) {
      const int n = Nbase + wn * 64 + ni * 16 + r;
#pragma unroll
      for (int i = 0; i < 8; ++i) {
        const int m = Mbase + wm * 32 + mi * 16 + i + hh;
        C[((size_t)bb * M + m) * 4096 + n] = acc[mi][ni][i];
      }
    }
}

// ---------------------------------------------------------------------------
// BatchNorm (train-mode batch stats).  x layout [b][C][N].
// ---------------------------------------------------------------------------
__global__ __launch_bounds__(256) void bn_stats_kernel(const float* __restrict__ x,
                                                       float* __restrict__ mean,
                                                       float* __restrict__ rstd,
                                                       int C, int N, int batch) {
  const int c = blockIdx.x;
  float s = 0.f, s2 = 0.f;
  for (int b = 0; b < batch; ++b) {
    const float* p = x + ((size_t)b * C + c) * N;
    for (int i = threadIdx.x; i < N; i += 256) {
      float v = p[i];
      s += v;
      s2 += v * v;
    }
  }
  __shared__ float rs[256], rq[256];
  rs[threadIdx.x] = s;
  rq[threadIdx.x] = s2;
  __syncthreads();
  for (int off = 128; off > 0; off >>= 1) {
    if (threadIdx.x < off) {
      rs[threadIdx.x] += rs[threadIdx.x + off];
      rq[threadIdx.x] += rq[threadIdx.x + off];
    }
    __syncthreads();
  }
  if (threadIdx.x == 0) {
    float inv = 1.f / (float)((size_t)N * batch);
    float m = rs[0] * inv;
    float var = rq[0] * inv - m * m;
    mean[c] = m;
    rstd[c] = rsqrtf(var + 1e-5f);
  }
}

__global__ __launch_bounds__(256) void bn_apply_kernel(const float* __restrict__ x,
                                                       const float* __restrict__ g,
                                                       const float* __restrict__ be,
                                                       const float* __restrict__ mean,
                                                       const float* __restrict__ rstd,
                                                       float* __restrict__ y32,
                                                       _Float16* __restrict__ y16,
                                                       int C, int N) {
  const int bc = blockIdx.x;  // b*C + c
  const int c = bc % C;
  const float sc = g[c] * rstd[c];
  const float sh = be[c] - mean[c] * sc;
  const float* p = x + (size_t)bc * N;
  for (int i = threadIdx.x; i < N; i += 256) {
    float v = fmaxf(p[i] * sc + sh, 0.f);
    if (y32) y32[(size_t)bc * N + i] = v;
    if (y16) y16[(size_t)bc * N + i] = (_Float16)v;
  }
}

// ---------------------------------------------------------------------------
// L = W_L @ low_level + b_L   -> [b][19][4096]
// ---------------------------------------------------------------------------
__global__ __launch_bounds__(256) void l_kernel(const float* __restrict__ low,
                                                const float* __restrict__ W_L,
                                                const float* __restrict__ b_L,
                                                float* __restrict__ L) {
  const int b = blockIdx.x >> 4;
  const int n = ((blockIdx.x & 15) << 8) + threadIdx.x;
  __shared__ float WLs[NC * 256];
  float acc[NC];
#pragma unroll
  for (int k = 0; k < NC; ++k) acc[k] = 0.f;
  for (int cc = 0; cc < 1024; cc += 256) {
    __syncthreads();
    for (int i = threadIdx.x; i < NC * 256; i += 256) {
      int k = i >> 8, c = i & 255;
      WLs[i] = W_L[(size_t)k * 1024 + cc + c];
    }
    __syncthreads();
    for (int c = 0; c < 256; ++c) {
      float xv = low[((size_t)b * 1024 + cc + c) * 4096 + n];
#pragma unroll
      for (int k = 0; k < NC; ++k) acc[k] += xv * WLs[(k << 8) + c];
    }
  }
#pragma unroll
  for (int k = 0; k < NC; ++k)
    L[((size_t)b * NC + k) * 4096 + n] = acc[k] + b_L[k];
}

// In-place softmax over 4096 per row (76 rows)
__global__ __launch_bounds__(256) void softmax_row_kernel(float* __restrict__ L) {
  float* row = L + (size_t)blockIdx.x * 4096;
  __shared__ float red[256];
  float m = -3.4e38f;
  for (int i = threadIdx.x; i < 4096; i += 256) m = fmaxf(m, row[i]);
  red[threadIdx.x] = m;
  __syncthreads();
  for (int off = 128; off > 0; off >>= 1) {
    if (threadIdx.x < off) red[threadIdx.x] = fmaxf(red[threadIdx.x], red[threadIdx.x + off]);
    __syncthreads();
  }
  m = red[0];
  __syncthreads();
  float s = 0.f;
  for (int i = threadIdx.x; i < 4096; i += 256) s += __expf(row[i] - m);
  red[threadIdx.x] = s;
  __syncthreads();
  for (int off = 128; off > 0; off >>= 1) {
    if (threadIdx.x < off) red[threadIdx.x] += red[threadIdx.x + off];
    __syncthreads();
  }
  const float inv = 1.f / red[0];
  __syncthreads();
  for (int i = threadIdx.x; i < 4096; i += 256) row[i] = __expf(row[i] - m) * inv;
}

// f_k[b][c][19] = sum_n M[b][k][n] * Xf[b][c][n]
__global__ __launch_bounds__(256) void fk_kernel(const float* __restrict__ Mx,
                                                 const _Float16* __restrict__ Xh,
                                                 float* __restrict__ FK) {
  const int b = blockIdx.x >> 1;
  const int c = ((blockIdx.x & 1) << 8) + threadIdx.x;
  __shared__ float Ms[NC * 256];
  float acc[NC];
#pragma unroll
  for (int k = 0; k < NC; ++k) acc[k] = 0.f;
  for (int nc = 0; nc < 4096; nc += 256) {
    __syncthreads();
    for (int i = threadIdx.x; i < NC * 256; i += 256) {
      int k = i >> 8, j = i & 255;
      Ms[i] = Mx[((size_t)b * NC + k) * 4096 + nc + j];
    }
    __syncthreads();
    for (int j = 0; j < 256; ++j) {
      float xv = (float)Xh[((size_t)b * 512 + c) * 4096 + nc + j];
#pragma unroll
      for (int k = 0; k < NC; ++k) acc[k] += xv * Ms[(k << 8) + j];
    }
  }
#pragma unroll
  for (int k = 0; k < NC; ++k) FK[((size_t)b * 512 + c) * NC + k] = acc[k];
}

// phi & delta raw: [b][256][19] = W @ f_k
__global__ __launch_bounds__(256) void qd_kernel(const float* __restrict__ FK,
                                                 const float* __restrict__ Wp,
                                                 const float* __restrict__ Wd,
                                                 float* __restrict__ Q,
                                                 float* __restrict__ D) {
  const int b = blockIdx.x / NC;
  const int k = blockIdx.x % NC;
  const int o = threadIdx.x;
  __shared__ float fks[512];
  for (int i = threadIdx.x; i < 512; i += 256) fks[i] = FK[((size_t)b * 512 + i) * NC + k];
  __syncthreads();
  float ap = 0.f, ad = 0.f;
  for (int c = 0; c < 512; ++c) {
    float fv = fks[c];
    ap += Wp[(size_t)o * 512 + c] * fv;
    ad += Wd[(size_t)o * 512 + c] * fv;
  }
  Q[((size_t)b * 256 + o) * NC + k] = ap;
  D[((size_t)b * 256 + o) * NC + k] = ad;
}

// BN+ReLU over the tiny [4][256][19] tensors, in-place (blockIdx 0=phi, 1=delta)
__global__ __launch_bounds__(256) void qd_bn_kernel(float* __restrict__ Q, float* __restrict__ D,
                                                    const float* gq, const float* bq,
                                                    const float* gd, const float* bd) {
  float* buf = blockIdx.x ? D : Q;
  const float* g = blockIdx.x ? gd : gq;
  const float* be = blockIdx.x ? bd : bq;
  const int o = threadIdx.x;
  float s = 0.f, s2 = 0.f;
  for (int b = 0; b < 4; ++b)
    for (int k = 0; k < NC; ++k) {
      float v = buf[((size_t)b * 256 + o) * NC + k];
      s += v;
      s2 += v * v;
    }
  const float inv = 1.f / (4.f * NC);
  const float m = s * inv;
  const float rs = rsqrtf(s2 * inv - m * m + 1e-5f);
  const float sc = g[o] * rs, sh = be[o] - m * sc;
  for (int b = 0; b < 4; ++b)
    for (int k = 0; k < NC; ++k) {
      size_t idx = ((size_t)b * 256 + o) * NC + k;
      buf[idx] = fmaxf(buf[idx] * sc + sh, 0.f);
    }
}

// logit[b][k][n] = sum_q query[b][q][k]*key[b][q][n]; softmax over k in registers.
__global__ __launch_bounds__(256) void attn_kernel(const float* __restrict__ Q,
                                                   const float* __restrict__ KEY,
                                                   float* __restrict__ ATTN) {
  const int b = blockIdx.x >> 4;
  const int n = ((blockIdx.x & 15) << 8) + threadIdx.x;
  __shared__ float qs[256 * NC];
  for (int i = threadIdx.x; i < 256 * NC; i += 256) qs[i] = Q[(size_t)b * 256 * NC + i];
  __syncthreads();
  float acc[NC];
#pragma unroll
  for (int k = 0; k < NC; ++k) acc[k] = 0.f;
  for (int o = 0; o < 256; ++o) {
    float kv = KEY[((size_t)b * 256 + o) * 4096 + n];
#pragma unroll
    for (int k = 0; k < NC; ++k) acc[k] += qs[o * NC + k] * kv;
  }
  float mx = acc[0];
#pragma unroll
  for (int k = 1; k < NC; ++k) mx = fmaxf(mx, acc[k]);
  float s = 0.f;
#pragma unroll
  for (int k = 0; k < NC; ++k) {
    acc[k] = __expf(acc[k] - mx);
    s += acc[k];
  }
  const float inv = 1.f / s;
#pragma unroll
  for (int k = 0; k < NC; ++k) ATTN[((size_t)b * NC + k) * 4096 + n] = acc[k] * inv;
}

// attn_sum[b][q][n] = sum_k delta[b][q][k]*attn[b][k][n]  -> f16 for WMMA
__global__ __launch_bounds__(256) void attnsum_kernel(const float* __restrict__ D,
                                                      const float* __restrict__ ATTN,
                                                      _Float16* __restrict__ OUT) {
  const int b = blockIdx.x >> 7;
  const int nt = (blockIdx.x & 127) << 5;
  __shared__ float ds[256 * NC];
  __shared__ float at[NC * 32];
  for (int i = threadIdx.x; i < 256 * NC; i += 256) ds[i] = D[(size_t)b * 256 * NC + i];
  for (int i = threadIdx.x; i < NC * 32; i += 256) {
    int k = i >> 5, j = i & 31;
    at[i] = ATTN[((size_t)b * NC + k) * 4096 + nt + j];
  }
  __syncthreads();
  const int nl = threadIdx.x & 31, qg = threadIdx.x >> 5;
  float av[NC];
#pragma unroll
  for (int k = 0; k < NC; ++k) av[k] = at[(k << 5) + nl];
  for (int qq = 0; qq < 32; ++qq) {
    const int q = qg * 32 + qq;
    float s = 0.f;
#pragma unroll
    for (int k = 0; k < NC; ++k) s += ds[q * NC + k] * av[k];
    OUT[((size_t)b * 256 + q) * 4096 + nt + nl] = (_Float16)s;
  }
}

// out[b][19][n] = W_out @ X_bar + b_out
__global__ __launch_bounds__(256) void out_kernel(const float* __restrict__ XB,
                                                  const float* __restrict__ W,
                                                  const float* __restrict__ bias,
                                                  float* __restrict__ OUT) {
  const int b = blockIdx.x >> 4;
  const int n = ((blockIdx.x & 15) << 8) + threadIdx.x;
  __shared__ float Wo[512 * NC];  // Wo[c*19+k]
  __shared__ float bs[NC];
  for (int i = threadIdx.x; i < 512 * NC; i += 256) {
    int c = i / NC, k = i - c * NC;
    Wo[i] = W[(size_t)k * 512 + c];
  }
  if (threadIdx.x < NC) bs[threadIdx.x] = bias[threadIdx.x];
  __syncthreads();
  float acc[NC];
#pragma unroll
  for (int k = 0; k < NC; ++k) acc[k] = bs[k];
  for (int c = 0; c < 512; ++c) {
    float xv = XB[((size_t)b * 512 + c) * 4096 + n];
#pragma unroll
    for (int k = 0; k < NC; ++k) acc[k] += Wo[c * NC + k] * xv;
  }
#pragma unroll
  for (int k = 0; k < NC; ++k) OUT[((size_t)b * NC + k) * 4096 + n] = acc[k];
}

// ---------------------------------------------------------------------------
extern "C" void kernel_launch(void* const* d_in, const int* in_sizes, int n_in,
                              void* d_out, int out_size, void* d_ws, size_t ws_size,
                              hipStream_t stream) {
  const float* x     = (const float*)d_in[0];
  const float* low   = (const float*)d_in[1];
  const float* W_L   = (const float*)d_in[2];
  const float* b_L   = (const float*)d_in[3];
  const float* W_X   = (const float*)d_in[4];
  const float* g_X   = (const float*)d_in[5];
  const float* be_X  = (const float*)d_in[6];
  const float* W_phi = (const float*)d_in[7];
  const float* g_phi = (const float*)d_in[8];
  const float* be_phi= (const float*)d_in[9];
  const float* W_psi = (const float*)d_in[10];
  const float* g_psi = (const float*)d_in[11];
  const float* be_psi= (const float*)d_in[12];
  const float* W_del = (const float*)d_in[13];
  const float* g_del = (const float*)d_in[14];
  const float* be_del= (const float*)d_in[15];
  const float* W_rho = (const float*)d_in[16];
  const float* g_rho = (const float*)d_in[17];
  const float* be_rho= (const float*)d_in[18];
  const float* W_g   = (const float*)d_in[19];
  const float* g_g   = (const float*)d_in[20];
  const float* be_g  = (const float*)d_in[21];
  const float* W_out = (const float*)d_in[22];
  const float* b_out = (const float*)d_in[23];
  float* out = (float*)d_out;

  char* ws = (char*)d_ws;
  // ---- workspace layout (offsets 256B aligned) ----
  const size_t o_xpad   = 0;                         // 77,856,768  f16 [4][2048][66][72]
  const size_t o_wxh    = o_xpad + 77856768;         // 18,874,368  f16
  const size_t o_xfraw  = o_wxh + 18874368;          // 33,554,432  f32 (reused as Xbar)
  const size_t o_xfh    = o_xfraw + 33554432;        // 16,777,216  f16
  const size_t o_xobjh  = o_xfh + 16777216;          // 16,777,216  f16
  const size_t o_wpsih  = o_xobjh + 16777216;        // 262,144
  const size_t o_wrhoh  = o_wpsih + 262144;          // 262,144
  const size_t o_wgh    = o_wrhoh + 262144;          // 1,048,576
  const size_t o_mean   = o_wgh + 1048576;           // 2,048
  const size_t o_rstd   = o_mean + 2048;             // 2,048
  // unions (dead regions reused after the conv):
  const size_t o_keyraw = o_xpad;                    // 16,777,216 f32
  const size_t o_attn   = o_xpad + 16777216;         // 1,245,184  f32
  const size_t o_asumh  = o_attn + 1245184;          // 8,388,608  f16
  const size_t o_xobjr  = o_asumh + 8388608;         // 33,554,432 f32
  const size_t o_L      = o_wxh;                     // 1,245,184  f32 (in-place softmax -> M)
  const size_t o_fk     = o_L + 1245184;             // 155,648
  const size_t o_qraw   = o_fk + 155648;             // 77,824
  const size_t o_draw   = o_qraw + 77824;            // 77,824

  _Float16* XPAD  = (_Float16*)(ws + o_xpad);
  _Float16* WXH   = (_Float16*)(ws + o_wxh);
  float*    XFRAW = (float*)(ws + o_xfraw);
  _Float16* XFH   = (_Float16*)(ws + o_xfh);
  _Float16* XOBJH = (_Float16*)(ws + o_xobjh);
  _Float16* WPSIH = (_Float16*)(ws + o_wpsih);
  _Float16* WRHOH = (_Float16*)(ws + o_wrhoh);
  _Float16* WGH   = (_Float16*)(ws + o_wgh);
  float*    MEAN  = (float*)(ws + o_mean);
  float*    RSTD  = (float*)(ws + o_rstd);
  float*    KEYR  = (float*)(ws + o_keyraw);
  float*    ATTN  = (float*)(ws + o_attn);
  _Float16* ASUMH = (_Float16*)(ws + o_asumh);
  float*    XOBJR = (float*)(ws + o_xobjr);
  float*    Lbuf  = (float*)(ws + o_L);
  float*    FK    = (float*)(ws + o_fk);
  float*    QRAW  = (float*)(ws + o_qraw);
  float*    DRAW  = (float*)(ws + o_draw);

  // 1) prep: pad x -> f16 (row pitch 72); weights -> f16 (k-contiguous)
  pad_x_kernel<<<(unsigned)(((size_t)4 * 2048 * 66 * 72 + 255) / 256), 256, 0, stream>>>(x, XPAD);
  wx_convert_kernel<<<(512 * 18432) / 256, 256, 0, stream>>>(W_X, WXH);
  cast_f16_kernel<<<(256 * 512) / 256, 256, 0, stream>>>(W_psi, WPSIH, 256 * 512);
  cast_f16_kernel<<<(512 * 256) / 256, 256, 0, stream>>>(W_rho, WRHOH, 512 * 256);
  cast_f16_kernel<<<(512 * 1024) / 256, 256, 0, stream>>>(W_g, WGH, 512 * 1024);

  // 2) 3x3 conv via implicit-GEMM WMMA; then BN(train)+ReLU -> f16 Xf
  conv_wmma_kernel<<<dim3(128, 4, 1), 256, 0, stream>>>(WXH, XPAD, XFRAW);
  bn_stats_kernel<<<512, 256, 0, stream>>>(XFRAW, MEAN, RSTD, 512, 4096, 4);
  bn_apply_kernel<<<4 * 512, 256, 0, stream>>>(XFRAW, g_X, be_X, MEAN, RSTD,
                                               (float*)nullptr, XFH, 512, 4096);

  // 3) soft object regions: L -> softmax over N -> M (in-place)
  l_kernel<<<4 * 16, 256, 0, stream>>>(low, W_L, b_L, Lbuf);
  softmax_row_kernel<<<4 * NC, 256, 0, stream>>>(Lbuf);

  // 4) region features f_k, then phi/delta (+tiny BN)
  fk_kernel<<<4 * 2, 256, 0, stream>>>(Lbuf, XFH, FK);
  qd_kernel<<<4 * NC, 256, 0, stream>>>(FK, W_phi, W_del, QRAW, DRAW);
  qd_bn_kernel<<<2, 256, 0, stream>>>(QRAW, DRAW, g_phi, be_phi, g_del, be_del);

  // 5) key = psi(Xf) via WMMA, BN+ReLU in place (f32)
  gemm_wmma_kernel<<<dim3(32, 2, 4), 256, 0, stream>>>(WPSIH, XFH, (const _Float16*)nullptr,
                                                       KEYR, 256, 512, 0);
  bn_stats_kernel<<<256, 256, 0, stream>>>(KEYR, MEAN, RSTD, 256, 4096, 4);
  bn_apply_kernel<<<4 * 256, 256, 0, stream>>>(KEYR, g_psi, be_psi, MEAN, RSTD,
                                               KEYR, (_Float16*)nullptr, 256, 4096);

  // 6) logits + softmax over nc (registers), then delta-weighted sum -> f16
  attn_kernel<<<4 * 16, 256, 0, stream>>>(QRAW, KEYR, ATTN);
  attnsum_kernel<<<4 * 128, 256, 0, stream>>>(DRAW, ATTN, ASUMH);

  // 7) X_obj = rho(attn_sum) via WMMA, BN+ReLU -> f16
  gemm_wmma_kernel<<<dim3(32, 4, 4), 256, 0, stream>>>(WRHOH, ASUMH, (const _Float16*)nullptr,
                                                       XOBJR, 512, 256, 0);
  bn_stats_kernel<<<512, 256, 0, stream>>>(XOBJR, MEAN, RSTD, 512, 4096, 4);
  bn_apply_kernel<<<4 * 512, 256, 0, stream>>>(XOBJR, g_rho, be_rho, MEAN, RSTD,
                                               (float*)nullptr, XOBJH, 512, 4096);

  // 8) X_bar = g([Xf ; X_obj]) via WMMA with split-K concat; BN+ReLU in place
  gemm_wmma_kernel<<<dim3(32, 4, 4), 256, 0, stream>>>(WGH, XFH, XOBJH,
                                                       XFRAW, 512, 512, 512);
  bn_stats_kernel<<<512, 256, 0, stream>>>(XFRAW, MEAN, RSTD, 512, 4096, 4);
  bn_apply_kernel<<<4 * 512, 256, 0, stream>>>(XFRAW, g_g, be_g, MEAN, RSTD,
                                               XFRAW, (_Float16*)nullptr, 512, 4096);

  // 9) final 19-channel 1x1 conv
  out_kernel<<<4 * 16, 256, 0, stream>>>(XFRAW, W_out, b_out, out);
  (void)in_sizes; (void)n_in; (void)out_size; (void)ws_size;
}
